// LuongSeq2Seq_62027917689187
// MI455X (gfx1250) — matrix operand order repaired and checked by reading, compiled, not run on hardware
//
#include <hip/hip_runtime.h>
#include <math.h>

// ---------------------------------------------------------------------------
// Types for WMMA
// ---------------------------------------------------------------------------
typedef __attribute__((ext_vector_type(16))) __bf16 v16bf;
typedef __attribute__((ext_vector_type(8)))  float  v8f;
typedef __attribute__((ext_vector_type(4)))  unsigned int v4u;

union FragBF {
    v4u   q[2];
    v16bf v;
};

// ---------------------------------------------------------------------------
// Elementwise / utility kernels
// ---------------------------------------------------------------------------
__global__ void gather_k(const int* __restrict__ tok, const float* __restrict__ emb,
                         float* __restrict__ out, long total, int H) {
    long i = (long)blockIdx.x * blockDim.x + threadIdx.x;
    if (i < total) {
        long t = i / H;
        int  k = (int)(i - t * H);
        out[i] = emb[(long)tok[t] * H + k];
    }
}

__global__ void cvt_bf16_k(const float* __restrict__ in, unsigned short* __restrict__ out, long n) {
    long i = (long)blockIdx.x * blockDim.x + threadIdx.x;
    if (i < n) {
        unsigned u = __float_as_uint(in[i]);
        unsigned r = u + 0x7FFFu + ((u >> 16) & 1u);   // round-to-nearest-even
        out[i] = (unsigned short)(r >> 16);
    }
}

__global__ void zero_k(float* __restrict__ p, long n) {
    long i = (long)blockIdx.x * blockDim.x + threadIdx.x;
    if (i < n) p[i] = 0.0f;
}

// ---------------------------------------------------------------------------
// Register-blocked WMMA bf16 GEMM:  C(M,N) = A(M,K) @ W(N,K)^T  [+ bias(N)]
// A, W bf16 row-major; C f32. M,N mult of 32, K mult of 32.
// Each wave owns a 32x32 output macro-tile: 2 A-frags x 2 B-frags -> 4
// v_wmma_f32_16x16x32_bf16 per K-step (16 FLOP/byte of fragment traffic,
// 4 independent accumulator chains for the matrix pipe).
// ---------------------------------------------------------------------------
__global__ void wmma_gemm_bf16(const unsigned short* __restrict__ A,
                               const unsigned short* __restrict__ W,
                               const float* __restrict__ bias,
                               float* __restrict__ C,
                               int M, int N, int K) {
    const int lane  = threadIdx.x & 31;
    const int wave  = threadIdx.x >> 5;
    const int stN   = N >> 5;                       // 32-wide supertiles
    const int stM   = M >> 5;
    const int st    = blockIdx.x * (blockDim.x >> 5) + wave;
    if (st >= stM * stN) return;                    // wave-uniform exit (EXEC all-ones)

    const int row0 = (st / stN) << 5;
    const int col0 = (st % stN) << 5;

    // A-matrix fragment (16x32 bf16): lanes<16 -> K 0..15, lanes>=16 -> K 16..31.
    const long aRow0 = (long)(row0 + (lane & 15)) * K + ((lane < 16) ? 0 : 16);
    const long aRow1 = aRow0 + (long)16 * K;
    // B-matrix fragment (32x16 bf16): lanes<16 -> K 0..15, lanes>=16 -> K 16..31.
    const long bRow0 = (long)(col0 + (lane & 15)) * K + ((lane < 16) ? 0 : 16);
    const long bRow1 = bRow0 + (long)16 * K;

    v8f acc00 = {}, acc01 = {}, acc10 = {}, acc11 = {};
    for (int k0 = 0; k0 < K; k0 += 32) {
        FragBF a0, a1, b0, b1;
        const v4u* pa0 = (const v4u*)(A + aRow0 + k0);
        a0.q[0] = pa0[0];  a0.q[1] = pa0[1];        // 8 consecutive dwords
        const v4u* pa1 = (const v4u*)(A + aRow1 + k0);
        a1.q[0] = pa1[0];  a1.q[1] = pa1[1];
        const v4u* pb0 = (const v4u*)(W + bRow0 + k0);
        b0.q[0] = pb0[0];  b0.q[1] = pb0[1];        // 8 consecutive dwords
        const v4u* pb1 = (const v4u*)(W + bRow1 + k0);
        b1.q[0] = pb1[0];  b1.q[1] = pb1[1];
        acc00 = __builtin_amdgcn_wmma_f32_16x16x32_bf16(false, a0.v, false, b0.v,
                                                        (short)0, acc00, false, false);
        acc01 = __builtin_amdgcn_wmma_f32_16x16x32_bf16(false, a0.v, false, b1.v,
                                                        (short)0, acc01, false, false);
        acc10 = __builtin_amdgcn_wmma_f32_16x16x32_bf16(false, a1.v, false, b0.v,
                                                        (short)0, acc10, false, false);
        acc11 = __builtin_amdgcn_wmma_f32_16x16x32_bf16(false, a1.v, false, b1.v,
                                                        (short)0, acc11, false, false);
    }

    // C/D layout: col = lane&15 (+16 for right tiles); rows r + 8*(lane>=16) (+16 lower).
    const int   cn0 = col0 + (lane & 15);
    const int   cn1 = cn0 + 16;
    const int   rb0 = row0 + ((lane < 16) ? 0 : 8);
    const int   rb1 = rb0 + 16;
    const float bv0 = bias ? bias[cn0] : 0.0f;
    const float bv1 = bias ? bias[cn1] : 0.0f;
#pragma unroll
    for (int r = 0; r < 8; ++r) {
        C[(long)(rb0 + r) * N + cn0] = acc00[r] + bv0;
        C[(long)(rb0 + r) * N + cn1] = acc01[r] + bv1;
        C[(long)(rb1 + r) * N + cn0] = acc10[r] + bv0;
        C[(long)(rb1 + r) * N + cn1] = acc11[r] + bv1;
    }
}

// ---------------------------------------------------------------------------
// Fused GRU step. gridDim.y==2 runs fwd (y=0) and bwd (y=1) directions
// (reference shares U weights between directions). Each thread produces one
// h_new[b][j], fusing the three h@U^T dot products with the gate math.
//   r = sigmoid(xp[0H+j] + h.Ur_j); z = sigmoid(xp[1H+j] + h.Uz_j)
//   n = tanh(xp[2H+j] + r*(h.Un_j + bn_j)); h' = (1-z)*n + z*h
// ---------------------------------------------------------------------------
__global__ void gru_step_k(const float* __restrict__ xp_f, const float* __restrict__ xp_b,
                           const float* __restrict__ hf_in, const float* __restrict__ hb_in,
                           float* __restrict__ hf_out, float* __restrict__ hb_out,
                           float* __restrict__ of, float* __restrict__ ob, int ostride,
                           const float* __restrict__ Ur, const float* __restrict__ Uz,
                           const float* __restrict__ Un, const float* __restrict__ bn,
                           int B, int H) {
    int idx = blockIdx.x * blockDim.x + threadIdx.x;
    if (idx >= B * H) return;
    int b = idx / H;
    int j = idx - b * H;

    const float* xp   = xp_f;
    const float* hin  = hf_in;
    float*       hout = hf_out;
    float*       o    = of;
    if (blockIdx.y) { xp = xp_b; hin = hb_in; hout = hb_out; o = ob; }

    const float* hr = hin + (long)b * H;
    const float* ur = Ur + (long)j * H;
    const float* uz = Uz + (long)j * H;
    const float* un = Un + (long)j * H;
    float dr = 0.f, dz = 0.f, dn = 0.f;
    for (int k = 0; k < H; ++k) {
        float hk = hr[k];
        dr = fmaf(hk, ur[k], dr);
        dz = fmaf(hk, uz[k], dz);
        dn = fmaf(hk, un[k], dn);
    }
    const float* xr = xp + (long)b * 3 * H;
    float r = 1.f / (1.f + expf(-(xr[j] + dr)));
    float z = 1.f / (1.f + expf(-(xr[H + j] + dz)));
    float n = tanhf(xr[2 * H + j] + r * (dn + bn[j]));
    float hnew = (1.f - z) * n + z * hr[j];
    hout[(long)b * H + j] = hnew;
    if (o) o[(long)b * ostride + j] = hnew;
}

// ---------------------------------------------------------------------------
// Generic small linear: out(B,N) = [inA(B,Ka) | inB(B,Kb)] @ W(N,Ka+Kb)^T (+bias) (opt tanh)
// ---------------------------------------------------------------------------
__global__ void linear2_k(const float* __restrict__ inA, int Ka,
                          const float* __restrict__ inB, int Kb,
                          const float* __restrict__ W, const float* __restrict__ bias,
                          float* __restrict__ out, int B, int N, int act) {
    int idx = blockIdx.x * blockDim.x + threadIdx.x;
    if (idx >= B * N) return;
    int b = idx / N;
    int j = idx - b * N;
    const float* w = W + (long)j * (Ka + Kb);
    float s = bias ? bias[j] : 0.0f;
    const float* a = inA + (long)b * Ka;
    for (int k = 0; k < Ka; ++k) s = fmaf(a[k], w[k], s);
    if (inB) {
        const float* bb = inB + (long)b * Kb;
        for (int k = 0; k < Kb; ++k) s = fmaf(bb[k], w[Ka + k], s);
    }
    if (act) s = tanhf(s);
    out[(long)b * N + j] = s;
}

// ---------------------------------------------------------------------------
// Attention: one block per batch element.
// scores[t] = scale * q[b].Henc[t][b], masked where src token == PAD(0);
// softmax over Ts; ctx[b] = sum_t alpha[t]*Henc[t][b].
// ---------------------------------------------------------------------------
__global__ void attn_k(const float* __restrict__ q, const float* __restrict__ Henc,
                       const int* __restrict__ src, float* __restrict__ ctx,
                       int B, int Ts, int H2, float scale) {
    int b = blockIdx.x;
    __shared__ float sc[64];
    for (int t = threadIdx.x; t < Ts; t += blockDim.x) {
        const float* hr = Henc + ((long)t * B + b) * H2;
        const float* qr = q + (long)b * H2;
        float s = 0.f;
        for (int k = 0; k < H2; ++k) s = fmaf(qr[k], hr[k], s);
        s *= scale;
        if (src[t * B + b] == 0) s = -1e9f;   // PAD mask
        sc[t] = s;
    }
    __syncthreads();
    if (threadIdx.x == 0) {
        float m = sc[0];
        for (int t = 1; t < Ts; ++t) m = fmaxf(m, sc[t]);
        float sum = 0.f;
        for (int t = 0; t < Ts; ++t) { float e = expf(sc[t] - m); sc[t] = e; sum += e; }
        float inv = 1.f / sum;
        for (int t = 0; t < Ts; ++t) sc[t] *= inv;
    }
    __syncthreads();
    for (int d = threadIdx.x; d < H2; d += blockDim.x) {
        float s = 0.f;
        for (int t = 0; t < Ts; ++t) s = fmaf(sc[t], Henc[((long)t * B + b) * H2 + d], s);
        ctx[(long)b * H2 + d] = s;
    }
}

// ---------------------------------------------------------------------------
// Host orchestration
// ---------------------------------------------------------------------------
extern "C" void kernel_launch(void* const* d_in, const int* in_sizes, int n_in,
                              void* d_out, int out_size, void* d_ws, size_t ws_size,
                              hipStream_t stream) {
    const int H = 512, V = 32000, TS = 48, TT = 48, B = 64;
    const int H2 = 2 * H, H3 = 3 * H;
    (void)in_sizes; (void)n_in; (void)out_size; (void)ws_size;

    const int*   src    = (const int*)d_in[0];
    const int*   tgt    = (const int*)d_in[1];
    const float* emb    = (const float*)d_in[2];
    const float* e0_Wih = (const float*)d_in[3];
    const float* e0_bih = (const float*)d_in[4];
    const float* e0_Ur  = (const float*)d_in[5];
    const float* e0_Uz  = (const float*)d_in[6];
    const float* e0_Un  = (const float*)d_in[7];
    const float* e0_bn  = (const float*)d_in[8];
    const float* e1_Wih = (const float*)d_in[9];
    const float* e1_bih = (const float*)d_in[10];
    const float* e1_Ur  = (const float*)d_in[11];
    const float* e1_Uz  = (const float*)d_in[12];
    const float* e1_Un  = (const float*)d_in[13];
    const float* e1_bn  = (const float*)d_in[14];
    const float* fc_w   = (const float*)d_in[15];
    const float* fc_b   = (const float*)d_in[16];
    const float* Wa     = (const float*)d_in[17];
    const float* Wc     = (const float*)d_in[18];
    const float* dW_ih  = (const float*)d_in[19];
    const float* dB_ih  = (const float*)d_in[20];
    const float* dU_r   = (const float*)d_in[21];
    const float* dU_z   = (const float*)d_in[22];
    const float* dU_n   = (const float*)d_in[23];
    const float* dB_n   = (const float*)d_in[24];
    const float* out_w  = (const float*)d_in[25];
    float*       logits = (float*)d_out;

    // Bump allocator on workspace (256B aligned chunks)
    char*  wsb = (char*)d_ws;
    size_t off = 0;
    auto alloc = [&](size_t bytes) -> void* {
        off = (off + 255) & ~(size_t)255;
        void* p = wsb + off;
        off += bytes;
        return p;
    };
    float* x_src  = (float*)alloc((size_t)TS * B * H * 4);
    float* tgt_e  = (float*)alloc((size_t)TT * B * H * 4);
    float* xp     = (float*)alloc((size_t)TS * B * H3 * 4);
    float* l0     = (float*)alloc((size_t)TS * B * H2 * 4);
    float* Henc   = (float*)alloc((size_t)TS * B * H2 * 4);
    float* hstA   = (float*)alloc((size_t)B * H * 4);
    float* hstB   = (float*)alloc((size_t)B * H * 4);
    float* hstC   = (float*)alloc((size_t)B * H * 4);
    float* hstD   = (float*)alloc((size_t)B * H * 4);
    float* decA   = (float*)alloc((size_t)B * H * 4);
    float* decB   = (float*)alloc((size_t)B * H * 4);
    float* ht0    = (float*)alloc((size_t)B * H * 4);
    float* dxp    = (float*)alloc((size_t)B * H3 * 4);
    float* qbuf   = (float*)alloc((size_t)B * H2 * 4);
    float* ctx    = (float*)alloc((size_t)B * H2 * 4);
    float* ht_all = (float*)alloc((size_t)TT * B * H * 4);
    unsigned short* Abf  = (unsigned short*)alloc((size_t)TS * B * H2 * 2); // up to 3072x1024
    unsigned short* w0bf = (unsigned short*)alloc((size_t)H3 * H * 2);
    unsigned short* w1bf = (unsigned short*)alloc((size_t)H3 * H2 * 2);
    unsigned short* owbf = (unsigned short*)alloc((size_t)V * H * 2);
    unsigned short* htbf = (unsigned short*)alloc((size_t)TT * B * H * 2);

    const int TB = 256;
    auto grid1 = [&](long n) { return dim3((unsigned)((n + TB - 1) / TB)); };
    auto launch_gemm = [&](const unsigned short* A, const unsigned short* W,
                           const float* bias, float* C, int M, int N, int K) {
        long tiles = (long)(M >> 5) * (N >> 5);    // 32x32 macro-tiles, 1 per wave
        wmma_gemm_bf16<<<dim3((unsigned)((tiles + 7) / 8)), 256, 0, stream>>>(A, W, bias, C, M, N, K);
    };

    // Embedding gathers
    gather_k<<<grid1((long)TS * B * H), TB, 0, stream>>>(src, emb, x_src, (long)TS * B * H, H);
    gather_k<<<grid1((long)TT * B * H), TB, 0, stream>>>(tgt, emb, tgt_e, (long)TT * B * H, H);

    // One-time bf16 weight conversions (recomputed every call; deterministic)
    cvt_bf16_k<<<grid1((long)H3 * H), TB, 0, stream>>>(e0_Wih, w0bf, (long)H3 * H);
    cvt_bf16_k<<<grid1((long)H3 * H2), TB, 0, stream>>>(e1_Wih, w1bf, (long)H3 * H2);
    cvt_bf16_k<<<grid1((long)V * H), TB, 0, stream>>>(out_w, owbf, (long)V * H);

    // ---- Encoder layer 0: batched input projection (WMMA), then recurrence ----
    cvt_bf16_k<<<grid1((long)TS * B * H), TB, 0, stream>>>(x_src, Abf, (long)TS * B * H);
    launch_gemm(Abf, w0bf, e0_bih, xp, TS * B, H3, H);

    zero_k<<<grid1((long)B * H), TB, 0, stream>>>(hstA, (long)B * H);
    zero_k<<<grid1((long)B * H), TB, 0, stream>>>(hstC, (long)B * H);
    float* hf_in = hstA; float* hf_out = hstB;
    float* hb_in = hstC; float* hb_out = hstD;
    dim3 gru_grid((unsigned)((B * H + TB - 1) / TB), 2);
    for (int s = 0; s < TS; ++s) {
        gru_step_k<<<gru_grid, TB, 0, stream>>>(
            xp + (long)s * B * H3, xp + (long)(TS - 1 - s) * B * H3,
            hf_in, hb_in, hf_out, hb_out,
            l0 + (long)s * B * H2, l0 + (long)(TS - 1 - s) * B * H2 + H, H2,
            e0_Ur, e0_Uz, e0_Un, e0_bn, B, H);
        float* t;
        t = hf_in; hf_in = hf_out; hf_out = t;
        t = hb_in; hb_in = hb_out; hb_out = t;
    }

    // ---- Encoder layer 1 (K = 2H) ----
    cvt_bf16_k<<<grid1((long)TS * B * H2), TB, 0, stream>>>(l0, Abf, (long)TS * B * H2);
    launch_gemm(Abf, w1bf, e1_bih, xp, TS * B, H3, H2);

    zero_k<<<grid1((long)B * H), TB, 0, stream>>>(hf_in, (long)B * H);
    zero_k<<<grid1((long)B * H), TB, 0, stream>>>(hb_in, (long)B * H);
    for (int s = 0; s < TS; ++s) {
        gru_step_k<<<gru_grid, TB, 0, stream>>>(
            xp + (long)s * B * H3, xp + (long)(TS - 1 - s) * B * H3,
            hf_in, hb_in, hf_out, hb_out,
            Henc + (long)s * B * H2, Henc + (long)(TS - 1 - s) * B * H2 + H, H2,
            e1_Ur, e1_Uz, e1_Un, e1_bn, B, H);
        float* t;
        t = hf_in; hf_in = hf_out; hf_out = t;
        t = hb_in; hb_in = hb_out; hb_out = t;
    }
    // final states live in hf_in / hb_in

    // h0 = tanh([hf, hb] @ fc_w^T + fc_b)
    linear2_k<<<grid1((long)B * H), TB, 0, stream>>>(hf_in, H, hb_in, H, fc_w, fc_b, decA, B, H, 1);
    zero_k<<<grid1((long)B * H), TB, 0, stream>>>(ht0, (long)B * H);

    // ---- Decoder loop (sequential; logits deferred) ----
    float* h_in = decA; float* h_out = decB;
    const float* ht_prev = ht0;
    const float scale = 0.03125f;  // 1/sqrt(2H) = 1/32
    for (int t = 0; t < TT; ++t) {
        // xp = [tgt_emb_t, ht_prev] @ dec_Wih^T + dec_bih
        linear2_k<<<grid1((long)B * H3), TB, 0, stream>>>(
            tgt_e + (long)t * B * H, H, ht_prev, H, dW_ih, dB_ih, dxp, B, H3, 0);
        // h = GRU(xp, h)
        gru_step_k<<<dim3((unsigned)((B * H + TB - 1) / TB), 1), TB, 0, stream>>>(
            dxp, nullptr, h_in, nullptr, h_out, nullptr, nullptr, nullptr, 0,
            dU_r, dU_z, dU_n, dB_n, B, H);
        // q = h @ Wa^T
        linear2_k<<<grid1((long)B * H2), TB, 0, stream>>>(
            h_out, H, nullptr, 0, Wa, nullptr, qbuf, B, H2, 0);
        // attention -> ctx
        attn_k<<<dim3(B), 256, 0, stream>>>(qbuf, Henc, src, ctx, B, TS, H2, scale);
        // ht = tanh([h, ctx] @ Wc^T)  (stored for batched logits GEMM)
        linear2_k<<<grid1((long)B * H), TB, 0, stream>>>(
            h_out, H, ctx, H2, Wc, nullptr, ht_all + (long)t * B * H, B, H, 1);
        ht_prev = ht_all + (long)t * B * H;
        float* tmp = h_in; h_in = h_out; h_out = tmp;
    }

    // ---- Batched output projection: logits(3072, 32000) = ht_all @ out_w^T ----
    cvt_bf16_k<<<grid1((long)TT * B * H), TB, 0, stream>>>(ht_all, htbf, (long)TT * B * H);
    launch_gemm(htbf, owbf, nullptr, logits, TT * B, V, H);
}
